// DLRKernel_67637144978495
// MI455X (gfx1250) — matrix operand order repaired and checked by reading, not compiled, not run on hardware
//
#include <hip/hip_runtime.h>
#include <math.h>

// ---- problem constants (reference: C=1, H=512, N=4096, L=8192) ----
#define HDIM 512
#define NDIM 4096
#define LDIM 8192

#define TILE_M 128   // output rows (h) per workgroup
#define TILE_L 128   // output cols (l) per workgroup
#define TILE_K 32    // k-step = WMMA f16 K depth
#define APAD   40    // LDS row pitch in halves (32 + 8 pad -> conflict-free b128)

#define INV_2PI 0.15915494309189535f

typedef __attribute__((ext_vector_type(16))) _Float16 v16h;
typedef __attribute__((ext_vector_type(8)))  _Float16 v8h;
typedef __attribute__((ext_vector_type(4)))  _Float16 v4h;
typedef __attribute__((ext_vector_type(2)))  _Float16 v2h;
typedef __attribute__((ext_vector_type(8)))  float    v8f;
typedef __attribute__((ext_vector_type(4)))  float    v4f;

#if __has_builtin(__builtin_amdgcn_fractf)
#define FRACT(x) __builtin_amdgcn_fractf(x)
#else
#define FRACT(x) ((x) - floorf(x))
#endif
// v_cos_f32: D = cos(2*pi * S)  (input in revolutions)
#if __has_builtin(__builtin_amdgcn_cosf)
#define COS2PI(x) __builtin_amdgcn_cosf(x)
#else
#define COS2PI(x) __cosf((x) * 6.283185307179586f)
#endif

// Stage one (A,S) tile pair into the given LDS double-buffer slot.
//  A: W[h0..h0+127, kk..kk+31] f32 -> f16, row-major [m][k]
//  S: S^T[l][k] = cos(Lambda[kk+k]*(l0+l)) f16, generated with a Chebyshev
//     recurrence along l (2 seed cosines + 6 FMAs per lambda).
__device__ __forceinline__ void stage_tiles(
    const float* __restrict__ W, const float* __restrict__ U,
    _Float16 (* __restrict__ Ab)[APAD], _Float16 (* __restrict__ Sb)[APAD],
    int h0, int l0, int kk, int tid)
{
    // ---- A tile: 128 rows * 8 float4 = 1024 b128 loads, 4 per thread ----
    #pragma unroll
    for (int i = 0; i < 4; ++i) {
        const int q   = i * 256 + tid;
        const int row = q >> 3;
        const int c4  = (q & 7) * 4;
        v4f w4 = *(const v4f*)(W + (size_t)(h0 + row) * NDIM + kk + c4);
        v4h h4 = { (_Float16)w4.x, (_Float16)w4.y,
                   (_Float16)w4.z, (_Float16)w4.w };
        *(v4h*)&Ab[row][c4] = h4;
    }

    // ---- S tile: thread handles k = {2kp, 2kp+1}, l = lgrp*8 .. +7 ----
    const int   kp   = tid & 15;
    const int   lgrp = tid >> 4;
    const float u0   = U[kk + 2 * kp];        // lambda/(2*pi) in [0,1)
    const float u1   = U[kk + 2 * kp + 1];
    const float lf   = (float)(l0 + lgrp * 8);

    const float t0 = 2.0f * COS2PI(u0);       // Chebyshev factor 2*cos(lambda)
    const float t1 = 2.0f * COS2PI(u1);

    float pa = COS2PI(FRACT(u0 * lf));        // cos(lambda0 * l)
    float ca = COS2PI(FRACT(u0 * (lf + 1.0f)));
    float pb = COS2PI(FRACT(u1 * lf));
    float cb = COS2PI(FRACT(u1 * (lf + 1.0f)));

    _Float16* dst = &Sb[lgrp * 8][2 * kp];
    *(v2h*)dst = v2h{ (_Float16)pa, (_Float16)pb }; dst += APAD;
    *(v2h*)dst = v2h{ (_Float16)ca, (_Float16)cb }; dst += APAD;
    #pragma unroll
    for (int i = 2; i < 8; ++i) {
        const float na = __builtin_fmaf(t0, ca, -pa);   // c[l+1]=2cosL*c[l]-c[l-1]
        const float nb = __builtin_fmaf(t1, cb, -pb);
        *(v2h*)dst = v2h{ (_Float16)na, (_Float16)nb }; dst += APAD;
        pa = ca; ca = na;
        pb = cb; cb = nb;
    }
}

__global__ __launch_bounds__(256)
void dlr_cos_gemm(const float* __restrict__ W,      // (512, 4096) f32
                  const float* __restrict__ Lambda, // (4096,) f32
                  float* __restrict__ out)          // (512, 8192) f32
{
    __shared__ _Float16 Alds[2][TILE_M][APAD];  // double-buffered W tile
    __shared__ _Float16 Slds[2][TILE_L][APAD];  // double-buffered S^T tile
    __shared__ float    Ulds[NDIM];             // Lambda / (2*pi)

    const int tid  = threadIdx.x;
    const int lane = tid & 31;
    const int wave = tid >> 5;

    const int l0 = blockIdx.x * TILE_L;
    const int h0 = blockIdx.y * TILE_M;

    for (int i = tid; i < NDIM; i += 256)
        Ulds[i] = Lambda[i] * INV_2PI;
    __syncthreads();

    // wave tiling: 2 (M) x 4 (L) waves; each wave -> 64x32 outputs = 4x2 WMMA tiles
    const int wm  = (wave >> 2) * 64;
    const int wl  = (wave & 3)  * 32;
    const int l15 = lane & 15;
    const int g   = lane >> 4;   // lane half-group (ISA fragment layouts)

    v8f acc[4][2];
    #pragma unroll
    for (int mt = 0; mt < 4; ++mt)
        #pragma unroll
        for (int nt = 0; nt < 2; ++nt)
            acc[mt][nt] = v8f{};

    // prologue: stage first tile into buffer 0
    stage_tiles(W, Ulds, Alds[0], Slds[0], h0, l0, 0, tid);

    int buf = 0;
    for (int kk = 0; kk < NDIM; kk += TILE_K) {
        __syncthreads();   // staging of Alds/Slds[buf] complete

        // stage next tile into the other buffer while we compute this one
        if (kk + TILE_K < NDIM)
            stage_tiles(W, Ulds, Alds[buf ^ 1], Slds[buf ^ 1],
                        h0, l0, kk + TILE_K, tid);

        // ---- B fragments (32x16 f16): N = l15, halves K {8g..8g+7,16+8g..23+8g} ----
        v16h bfrag[2];
        #pragma unroll
        for (int nt = 0; nt < 2; ++nt) {
            const _Float16* br = &Slds[buf][wl + nt * 16 + l15][0];
            v8h blo = *(const v8h*)(br + 8 * g);
            v8h bhi = *(const v8h*)(br + 16 + 8 * g);
            v16h b;
            #pragma unroll
            for (int i = 0; i < 8; ++i) { b[i] = blo[i]; b[i + 8] = bhi[i]; }
            bfrag[nt] = b;
        }

        // ---- A fragments (16x32 f16): M = l15, halves K {8g..8g+7,16+8g..23+8g} ----
        #pragma unroll
        for (int mt = 0; mt < 4; ++mt) {
            const _Float16* ar = &Alds[buf][wm + mt * 16 + l15][0];
            v8h alo = *(const v8h*)(ar + 8 * g);
            v8h ahi = *(const v8h*)(ar + 16 + 8 * g);
            v16h a;
            #pragma unroll
            for (int i = 0; i < 8; ++i) { a[i] = alo[i]; a[i + 8] = ahi[i]; }

            #pragma unroll
            for (int nt = 0; nt < 2; ++nt)
                acc[mt][nt] = __builtin_amdgcn_wmma_f32_16x16x32_f16(
                    false, a, false, bfrag[nt],
                    (short)0, acc[mt][nt], false, false);
        }

        buf ^= 1;
    }

    // ---- epilogue: C/D layout M = r + 8g, N = l15 ----
    #pragma unroll
    for (int mt = 0; mt < 4; ++mt) {
        #pragma unroll
        for (int nt = 0; nt < 2; ++nt) {
            const int l     = l0 + wl + nt * 16 + l15;
            const int hbase = h0 + wm + mt * 16 + 8 * g;
            #pragma unroll
            for (int r = 0; r < 8; ++r)
                out[(size_t)(hbase + r) * LDIM + l] = acc[mt][nt][r];
        }
    }
}

extern "C" void kernel_launch(void* const* d_in, const int* in_sizes, int n_in,
                              void* d_out, int out_size, void* d_ws, size_t ws_size,
                              hipStream_t stream) {
    const float* W      = (const float*)d_in[0];  // (1,512,4096) f32
    const float* Lambda = (const float*)d_in[1];  // (4096,) f32
    float* out = (float*)d_out;                   // (1,512,8192) f32

    dim3 grid(LDIM / TILE_L, HDIM / TILE_M);      // (64, 4) = 256 WGs
    dlr_cos_gemm<<<grid, 256, 0, stream>>>(W, Lambda, out);
}
